// Attention_68101001445614
// MI455X (gfx1250) — matrix operand order repaired and checked by reading, not compiled, not run on hardware
//
#include <hip/hip_runtime.h>
#include <hip/hip_bf16.h>

// ---------------------------------------------------------------------------
// MI455X (gfx1250) causal attention, bf16 WMMA pipeline.
// Stages: cvt(x,W) -> proj(Q,K,Vt) -> flash-attn (online softmax; Q j-tiles
// staged by the Tensor Data Mover, V j-tiles by async global->LDS loads,
// double-buffered in LDS) -> out-proj.
// Matrix math via v_wmma_f32_16x16x32_bf16 (f32 accumulate).
// ---------------------------------------------------------------------------

typedef __bf16 bf16;
typedef __attribute__((ext_vector_type(16))) __bf16 v16bf;
typedef __attribute__((ext_vector_type(8)))  __bf16 v8bf;   // 16 bytes
typedef __attribute__((ext_vector_type(8)))  float  v8f;
typedef __attribute__((ext_vector_type(4)))  unsigned v4u;
typedef __attribute__((ext_vector_type(8)))  int      v8i;
typedef __attribute__((ext_vector_type(4)))  int      v4i;

#define N_HEADS 8
#define EMB     256
#define KQ      64
#define BATCH   4
#define SEQ     2048

__device__ __forceinline__ bf16 f2bf(float f) {
  // round-to-nearest-even f32 -> bf16
  unsigned u = __builtin_bit_cast(unsigned, f);
  unsigned r = u + 0x7FFFu + ((u >> 16) & 1u);
  unsigned short h = (unsigned short)(r >> 16);
  return __builtin_bit_cast(bf16, h);
}

__device__ __forceinline__ v8f vzero8() {
  v8f z = {0.f, 0.f, 0.f, 0.f, 0.f, 0.f, 0.f, 0.f};
  return z;
}

// A fragment, 16x32 bf16 (MxK), row-major source with leading dim `ld`.
// ISA layout: lane L holds row (L&15); K chunks [8h, 8h+8) and [16+8h, 16+8h+8).
__device__ __forceinline__ v16bf load_a16x32(const bf16* base, int ld, int row0, int k0) {
  const int lane = threadIdx.x & 31;
  const int r = lane & 15, h = lane >> 4;
  const bf16* p = base + (size_t)(row0 + r) * ld + k0 + 8 * h;
  union { v16bf v; v8bf q[2]; } u;
  u.q[0] = *(const v8bf*)p;         // K = 8h .. 8h+7
  u.q[1] = *(const v8bf*)(p + 16);  // K = 16+8h .. 16+8h+7
  return u.v;
}

// B fragment, 32x16 bf16 (KxN), from a TRANSPOSED row-major source:
// src[col of B][k] with leading dim `ld`.  ISA layout: lane holds column
// (lane&15); K chunks [8h, 8h+8) and [16+8h, 16+8h+8).
__device__ __forceinline__ v16bf load_bT16x32(const bf16* base, int ld, int col0, int k0) {
  const int lane = threadIdx.x & 31;
  const int c = lane & 15, h = lane >> 4;
  const bf16* p = base + (size_t)(col0 + c) * ld + k0 + 8 * h;
  union { v16bf v; v8bf q[2]; } u;
  u.q[0] = *(const v8bf*)p;
  u.q[1] = *(const v8bf*)(p + 16);
  return u.v;
}

__device__ __forceinline__ v8f wmma_bf16(v16bf a, v16bf b, v8f c) {
  return __builtin_amdgcn_wmma_f32_16x16x32_bf16(
      /*neg_a=*/false, a, /*neg_b=*/false, b,
      /*c_mod=*/(short)0, c, /*reuse_a=*/false, /*reuse_b=*/false);
}

// ---- CDNA5 async global -> LDS (ASYNCcnt) ---------------------------------
__device__ __forceinline__ unsigned lds_addr_of(const void* p) {
  // generic pointers to LDS carry the byte offset in the low 32 bits
  return (unsigned)(uintptr_t)p;
}

__device__ __forceinline__ void async_ld16(unsigned lds_dst, const bf16* g) {
  asm volatile("global_load_async_to_lds_b128 %0, %1, off"
               :: "v"(lds_dst), "v"(g) : "memory");
}

__device__ __forceinline__ void wait_async0() {
  asm volatile("s_wait_asynccnt 0" ::: "memory");
}

// ---- CDNA5 Tensor Data Mover (TENSORcnt) ----------------------------------
// 2-D tile of 2-byte elements, tensor dims == tile dims (no OOB), row stride
// in elements.  D# bitfields per CDNA5 ISA ch.8 (groups 0/1; groups 2/3 zero
// for <=2D tensors).
__device__ __forceinline__ void tdm_load_2d(unsigned lds_addr, const void* gptr,
                                            unsigned d0, unsigned d1,
                                            unsigned stride_elems) {
  const unsigned long long ga = (unsigned long long)(uintptr_t)gptr;
  v4u g0;
  g0.x = 1u;                                  // count=1 (valid), user mode
  g0.y = lds_addr;                            // lds_addr (bytes)
  g0.z = (unsigned)(ga & 0xFFFFFFFFu);        // global_addr[31:0]
  g0.w = (unsigned)((ga >> 32) & 0x01FFFFFFu) // global_addr[56:32]
         | 0x80000000u;                       // type=2 ("image")
  v8i g1;
  g1[0] = 0x10000;                                        // data_size=1 (2B)
  g1[1] = (int)((d0 & 0xFFFFu) << 16);                    // tensor_dim0 lo
  g1[2] = (int)(((d0 >> 16) & 0xFFFFu) |
                ((d1 & 0xFFFFu) << 16));                  // td0 hi | td1 lo
  g1[3] = (int)(((d1 >> 16) & 0xFFFFu) |
                ((d0 & 0xFFFFu) << 16));                  // td1 hi | tile_dim0
  g1[4] = (int)(d1 & 0xFFFFu);                            // tile_dim1 (dim2=0)
  g1[5] = (int)stride_elems;                              // dim0 stride lo
  g1[6] = 0;                                              // stride0 hi | stride1 lo
  g1[7] = 0;
  const v4i gz = {0, 0, 0, 0};
#if defined(__clang_major__) && (__clang_major__ >= 23)
  const v8i gz8 = {0, 0, 0, 0, 0, 0, 0, 0};
  __builtin_amdgcn_tensor_load_to_lds(g0, g1, gz, gz, gz8, 0);
#else
  __builtin_amdgcn_tensor_load_to_lds(g0, g1, gz, gz, 0);
#endif
}

__device__ __forceinline__ void wait_tensor0() {
  __builtin_amdgcn_s_wait_tensorcnt((short)0);
}

// ---------------------------------------------------------------------------
// Conversion kernels
// ---------------------------------------------------------------------------
__global__ void k_cvt_x(const float* __restrict__ x, bf16* __restrict__ xb, int n) {
  int i = blockIdx.x * blockDim.x + threadIdx.x;
  if (i < n) xb[i] = f2bf(x[i]);
}

// in: [N][R][C] f32  ->  out: [N][C][R] bf16 (transpose inner 2 dims)
__global__ void k_cvt_T(const float* __restrict__ in, bf16* __restrict__ out,
                        int R, int C, int total) {
  int i = blockIdx.x * blockDim.x + threadIdx.x;
  if (i >= total) return;
  int n = i / (R * C);
  int rem = i - n * R * C;
  int r = rem / C;
  int c = rem - r * C;
  out[((size_t)n * C + c) * R + r] = f2bf(in[i]);
}

// ---------------------------------------------------------------------------
// Projection: Out[n][b][s][d] = sum_e xb[b][s][e] * W[n][e][d]
// wt is W transposed: [n][d][e].  transpose_out=1 stores Out as [n][b][d][s].
// One wave per 16x16 output tile; 8 WMMAs over E.
// ---------------------------------------------------------------------------
__global__ void k_proj(const bf16* __restrict__ xb, const bf16* __restrict__ wt,
                       bf16* __restrict__ out, int transpose_out) {
  int wave = (blockIdx.x * blockDim.x + threadIdx.x) >> 5;
  const int DT = KQ / 16, ST = SEQ / 16;
  const int dt = wave % DT; wave /= DT;
  const int st = wave % ST; wave /= ST;
  const int b  = wave % BATCH; wave /= BATCH;
  const int n  = wave;
  if (n >= N_HEADS) return;  // never taken (exact grid), keeps EXEC full

  const bf16* xa = xb + (size_t)b * SEQ * EMB;
  const bf16* wb = wt + (size_t)n * KQ * EMB;
  const int s0 = st * 16, d0 = dt * 16;

  v8f c = vzero8();
#pragma unroll
  for (int ek = 0; ek < EMB; ek += 32) {
    v16bf a  = load_a16x32(xa, EMB, s0, ek);
    v16bf bb = load_bT16x32(wb, EMB, d0, ek);
    c = wmma_bf16(a, bb, c);
  }

  const int lane = threadIdx.x & 31;
  const int colc = lane & 15, h = lane >> 4;
  if (!transpose_out) {
    bf16* o = out + ((size_t)n * BATCH + b) * SEQ * KQ;
#pragma unroll
    for (int r = 0; r < 8; ++r)
      o[(size_t)(s0 + r + 8 * h) * KQ + d0 + colc] = f2bf(c[r]);
  } else {
    bf16* o = out + ((size_t)n * BATCH + b) * KQ * SEQ;  // [d][s]
#pragma unroll
    for (int r = 0; r < 8; ++r)
      o[(size_t)(d0 + colc) * SEQ + s0 + r + 8 * h] = f2bf(c[r]);
  }
}

// ---------------------------------------------------------------------------
// Flash attention per (n,b): scores S[i,j] = K_i . Q_j, causal j<=i,
// softmax over j, ctx = P . V.
// One workgroup (8 waves) owns a 128-row i-block; each wave one 16-row tile.
// Per j-step the shared Q tile (32x64) is DMA'd by the TDM (wave 0 issues the
// descriptor), and the shared V tile (Vt rows, 64x32) is staged with async
// global->LDS loads by all 256 threads; both double-buffered.
// Km/Qm: [s][64]; Vt: [d][s].
// ---------------------------------------------------------------------------
#define ATTN_WAVES 8

__device__ __forceinline__ void stage_v_async(const bf16* __restrict__ Vp,
                                              int j0, bf16 (*Vs)[32], int tid) {
  // Vt rows d=0..63, cols j0..j0+31 (64B/row): 64 rows x 4 chunks of 16B
  const int row = tid >> 2, ch = tid & 3;
  async_ld16(lds_addr_of(&Vs[row][ch * 8]),
             Vp + (size_t)row * SEQ + j0 + ch * 8);
}

__global__ void __launch_bounds__(256)
k_attn(const bf16* __restrict__ Km, const bf16* __restrict__ Qm,
       const bf16* __restrict__ Vt, bf16* __restrict__ ctxb) {
  __shared__ __align__(16) bf16 Qs[2][32][64];             // 8 KB
  __shared__ __align__(16) bf16 Vs[2][64][32];             // 8 KB
  __shared__ __align__(16) bf16 plds[ATTN_WAVES][16][32];  // 8 KB

  const int tid   = threadIdx.x;
  const int wslot = tid >> 5;
  const int lane  = tid & 31;
  const int colc  = lane & 15, h = lane >> 4;

  int blk = blockIdx.x;
  const int IB = SEQ / 128;
  const int ib = blk % IB; blk /= IB;
  const int b  = blk % BATCH; blk /= BATCH;
  const int n  = blk;
  if (n >= N_HEADS) return;  // never taken

  const size_t nb = (size_t)n * BATCH + b;
  const bf16* Kp = Km + nb * SEQ * KQ;
  const bf16* Qp = Qm + nb * SEQ * KQ;
  const bf16* Vp = Vt + nb * KQ * SEQ;
  bf16*       Cp = ctxb + nb * SEQ * KQ;

  const int i0 = ib * 128 + wslot * 16;  // this wave's 16 rows
  const v16bf aK0 = load_a16x32(Kp, KQ, i0, 0);
  const v16bf aK1 = load_a16x32(Kp, KQ, i0, 32);

  v8f acc0 = vzero8(), acc1 = vzero8(), acc2 = vzero8(), acc3 = vzero8();
  float m[8], l[8];
#pragma unroll
  for (int r = 0; r < 8; ++r) { m[r] = -3.0e38f; l[r] = 0.f; }

  const int nj = (ib + 1) * 4;  // 32-wide j blocks up to the block's last row

  // prologue: stage j-block 0 into buffer 0
  stage_v_async(Vp, 0, Vs[0], tid);
  if (wslot == 0)
    tdm_load_2d(lds_addr_of(&Qs[0][0][0]), Qp, /*d0=*/KQ, /*d1=*/32, /*stride=*/KQ);

  for (int jb = 0; jb < nj; ++jb) {
    const int j0  = jb * 32;
    const int cur = jb & 1;

    wait_async0();                    // own V async loads done
    if (wslot == 0) wait_tensor0();   // Q TDM transfer done
    __syncthreads();                  // tiles visible; prev buffer free
    if (jb + 1 < nj) {                // prefetch next j-tile into other buffer
      stage_v_async(Vp, j0 + 32, Vs[cur ^ 1], tid);
      if (wslot == 0)
        tdm_load_2d(lds_addr_of(&Qs[cur ^ 1][0][0]),
                    Qp + (size_t)(j0 + 32) * KQ, KQ, 32, KQ);
    }

    const bf16* qs = &Qs[cur][0][0];
    const bf16* vs = &Vs[cur][0][0];

    // ---- scores: two 16x16 tiles (cols j0..j0+15, j0+16..j0+31), K=d=64
    v8f s0 = vzero8(), s1 = vzero8();
    s0 = wmma_bf16(aK0, load_bT16x32(qs, KQ, 0, 0), s0);
    s0 = wmma_bf16(aK1, load_bT16x32(qs, KQ, 0, 32), s0);
    s1 = wmma_bf16(aK0, load_bT16x32(qs, KQ, 16, 0), s1);
    s1 = wmma_bf16(aK1, load_bT16x32(qs, KQ, 16, 32), s1);

    // ---- causal mask + online softmax (row reductions across 16-lane halves)
    float p0[8], p1[8], scale[8];
#pragma unroll
    for (int r = 0; r < 8; ++r) {
      const int rowg = i0 + r + 8 * h;
      float v0 = ((j0 + colc)      <= rowg) ? s0[r] : -3.0e38f;
      float v1 = ((j0 + 16 + colc) <= rowg) ? s1[r] : -3.0e38f;
      float mx = fmaxf(v0, v1);
      mx = fmaxf(mx, __shfl_xor(mx, 1));
      mx = fmaxf(mx, __shfl_xor(mx, 2));
      mx = fmaxf(mx, __shfl_xor(mx, 4));
      mx = fmaxf(mx, __shfl_xor(mx, 8));
      const float mn = fmaxf(m[r], mx);
      scale[r] = __expf(m[r] - mn);
      p0[r] = __expf(v0 - mn);
      p1[r] = __expf(v1 - mn);
      float sum = p0[r] + p1[r];
      sum += __shfl_xor(sum, 1);
      sum += __shfl_xor(sum, 2);
      sum += __shfl_xor(sum, 4);
      sum += __shfl_xor(sum, 8);
      l[r] = l[r] * scale[r] + sum;
      m[r] = mn;
    }
#pragma unroll
    for (int r = 0; r < 8; ++r) {
      acc0[r] *= scale[r]; acc1[r] *= scale[r];
      acc2[r] *= scale[r]; acc3[r] *= scale[r];
    }

    // ---- re-layout P (C layout -> A layout) through per-wave LDS tile
#pragma unroll
    for (int r = 0; r < 8; ++r) {
      plds[wslot][r + 8 * h][colc]      = f2bf(p0[r]);
      plds[wslot][r + 8 * h][colc + 16] = f2bf(p1[r]);
    }
    __builtin_amdgcn_wave_barrier();  // same-wave LDS ops are in-order
    const v16bf aP = load_a16x32(&plds[wslot][0][0], 32, 0, 0);
    __builtin_amdgcn_wave_barrier();

    // ---- ctx += P(16x32) . V(32x64): 4 WMMAs over the d tiles
    acc0 = wmma_bf16(aP, load_bT16x32(vs, 32, 0,  0), acc0);
    acc1 = wmma_bf16(aP, load_bT16x32(vs, 32, 16, 0), acc1);
    acc2 = wmma_bf16(aP, load_bT16x32(vs, 32, 32, 0), acc2);
    acc3 = wmma_bf16(aP, load_bT16x32(vs, 32, 48, 0), acc3);
  }

  // ---- normalize and store ctx (bf16, [s][64])
#pragma unroll
  for (int r = 0; r < 8; ++r) {
    const float inv = 1.0f / l[r];
    const size_t row = (size_t)(i0 + r + 8 * h) * KQ;
    Cp[row + 0  + colc] = f2bf(acc0[r] * inv);
    Cp[row + 16 + colc] = f2bf(acc1[r] * inv);
    Cp[row + 32 + colc] = f2bf(acc2[r] * inv);
    Cp[row + 48 + colc] = f2bf(acc3[r] * inv);
  }
}

// ---------------------------------------------------------------------------
// Output projection + head reduction:
// out[b][s][e] = sum_n sum_d ctx[n][b][s][d] * Vout[n][d][e]
// voutt is Vout transposed: [n][e][d].  One wave per 16x16 out tile.
// ---------------------------------------------------------------------------
__global__ void k_out(const bf16* __restrict__ ctxb, const bf16* __restrict__ voutt,
                      float* __restrict__ out) {
  const int lane = threadIdx.x & 31;
  const int colc = lane & 15, h = lane >> 4;
  int wave = (blockIdx.x * blockDim.x + threadIdx.x) >> 5;
  const int ET = EMB / 16, ST = SEQ / 16;
  const int et = wave % ET; wave /= ET;
  const int st = wave % ST; wave /= ST;
  const int b  = wave;
  if (b >= BATCH) return;  // never taken

  const int s0 = st * 16, e0 = et * 16;
  v8f c = vzero8();
#pragma unroll
  for (int n = 0; n < N_HEADS; ++n) {
    const bf16* Cpn = ctxb + ((size_t)n * BATCH + b) * SEQ * KQ;
    const bf16* Von = voutt + (size_t)n * EMB * KQ;
    c = wmma_bf16(load_a16x32(Cpn, KQ, s0, 0),  load_bT16x32(Von, KQ, e0, 0),  c);
    c = wmma_bf16(load_a16x32(Cpn, KQ, s0, 32), load_bT16x32(Von, KQ, e0, 32), c);
  }
  float* o = out + (size_t)b * SEQ * EMB;
#pragma unroll
  for (int r = 0; r < 8; ++r)
    o[(size_t)(s0 + r + 8 * h) * EMB + e0 + colc] = c[r];
}

// ---------------------------------------------------------------------------
extern "C" void kernel_launch(void* const* d_in, const int* in_sizes, int n_in,
                              void* d_out, int out_size, void* d_ws, size_t ws_size,
                              hipStream_t stream) {
  (void)in_sizes; (void)n_in; (void)out_size; (void)ws_size;
  const float* x  = (const float*)d_in[0];
  const float* wk = (const float*)d_in[1];  // key_matrices   [n][e][d]
  const float* wq = (const float*)d_in[2];  // query_matrices [n][e][d]
  const float* wv = (const float*)d_in[3];  // value_in       [n][e][d]
  const float* wo = (const float*)d_in[4];  // value_out      [n][d][e]
  float* out = (float*)d_out;

  char* ws = (char*)d_ws;
  size_t off = 0;
  auto alloc = [&](size_t bytes) -> char* {
    char* p = ws + off;
    off += (bytes + 255) & ~(size_t)255;
    return p;
  };
  bf16* xb  = (bf16*)alloc((size_t)BATCH * SEQ * EMB * 2);        // 4 MB
  bf16* wqt = (bf16*)alloc((size_t)N_HEADS * KQ * EMB * 2);       // [n][d][e]
  bf16* wkt = (bf16*)alloc((size_t)N_HEADS * KQ * EMB * 2);
  bf16* wvt = (bf16*)alloc((size_t)N_HEADS * KQ * EMB * 2);
  bf16* wot = (bf16*)alloc((size_t)N_HEADS * EMB * KQ * 2);       // [n][e][d]
  bf16* Qm  = (bf16*)alloc((size_t)N_HEADS * BATCH * SEQ * KQ * 2);  // [n][b][s][d]
  bf16* Km  = (bf16*)alloc((size_t)N_HEADS * BATCH * SEQ * KQ * 2);
  bf16* Vt  = (bf16*)alloc((size_t)N_HEADS * BATCH * KQ * SEQ * 2);  // [n][b][d][s]
  bf16* Cb  = (bf16*)alloc((size_t)N_HEADS * BATCH * SEQ * KQ * 2);  // ~37 MB total

  // --- conversions
  {
    int n = BATCH * SEQ * EMB;
    k_cvt_x<<<(n + 255) / 256, 256, 0, stream>>>(x, xb, n);
  }
  {
    int n = N_HEADS * EMB * KQ;
    k_cvt_T<<<(n + 255) / 256, 256, 0, stream>>>(wq, wqt, EMB, KQ, n);
    k_cvt_T<<<(n + 255) / 256, 256, 0, stream>>>(wk, wkt, EMB, KQ, n);
    k_cvt_T<<<(n + 255) / 256, 256, 0, stream>>>(wv, wvt, EMB, KQ, n);
    k_cvt_T<<<(n + 255) / 256, 256, 0, stream>>>(wo, wot, KQ, EMB, n);
  }

  // --- projections: 16384 tiles, 4 waves/block
  const int proj_blocks = (N_HEADS * BATCH * (SEQ / 16) * (KQ / 16)) / 4;
  k_proj<<<proj_blocks, 128, 0, stream>>>(xb, wqt, Qm, 0);
  k_proj<<<proj_blocks, 128, 0, stream>>>(xb, wkt, Km, 0);
  k_proj<<<proj_blocks, 128, 0, stream>>>(xb, wvt, Vt, 1);

  // --- flash attention: one block per (n,b,128-row i-block)
  const int attn_blocks = N_HEADS * BATCH * (SEQ / 128);  // 512
  k_attn<<<attn_blocks, ATTN_WAVES * 32, 0, stream>>>(Km, Qm, Vt, Cb);

  // --- output projection: 8192 waves, 8 waves/block
  const int out_blocks = (BATCH * (SEQ / 16) * (EMB / 16)) / 8;
  k_out<<<out_blocks, 256, 0, stream>>>(Cb, wot, out);
}
